// MultiHeadRelationAwareAttention_16956530885254
// MI455X (gfx1250) — matrix operand order repaired and checked by reading, not compiled, not run on hardware
//
#include <hip/hip_runtime.h>

// ---------------------------------------------------------------------------
// MI455X (gfx1250) fused multi-head cross-batch attention, bf16 WMMA pipeline.
//   B=4096, D_MODEL=512, H=8, Dk=64
// ---------------------------------------------------------------------------

typedef __attribute__((ext_vector_type(16))) __bf16 bf16x16;
typedef __attribute__((ext_vector_type(8)))  __bf16 bf16x8;
typedef __attribute__((ext_vector_type(8)))  float  f32x8;

union ABFrag { bf16x16 v; bf16x8 h[2]; };

__device__ __forceinline__ f32x8 wmma_bf16(bf16x16 a, bf16x16 b, f32x8 c) {
  // D = A(16x32 bf16) * B(32x16 bf16) + C(16x16 f32)
  return __builtin_amdgcn_wmma_f32_16x16x32_bf16(
      /*neg_a=*/false, a, /*neg_b=*/false, b,
      /*c_mod=*/(short)0, c, /*reuse_a=*/false, /*reuse_b=*/false);
}

__device__ __forceinline__ unsigned short f2bf(float f) {
  unsigned int u = __float_as_uint(f);
  u += 0x7FFFu + ((u >> 16) & 1u);      // round-to-nearest-even
  return (unsigned short)(u >> 16);
}

// ---------------------------------------------------------------------------
// fp32 -> bf16 conversion
// ---------------------------------------------------------------------------
__global__ void cvt_bf16(const float* __restrict__ in,
                         unsigned short* __restrict__ out, int n) {
  int i = (blockIdx.x * blockDim.x + threadIdx.x) * 4;
  if (i + 3 < n) {
    float4 v = *(const float4*)(in + i);
    out[i + 0] = f2bf(v.x);
    out[i + 1] = f2bf(v.y);
    out[i + 2] = f2bf(v.z);
    out[i + 3] = f2bf(v.w);
  } else {
    for (; i < n; ++i) out[i] = f2bf(in[i]);
  }
}

// ---------------------------------------------------------------------------
// C[M x 512] = A[M x 512] @ W[512 x 512]^T + bias   (torch Linear semantics)
// MODE 0: store bf16 row-major (ldo = 512)
// MODE 1: store bf16 TRANSPOSED, Out[n*ldo + m]  (for V^T, ldo = 4096)
// MODE 2: store fp32 row-major (final output)
// Wave: 16 rows x 64 cols (A-frag reused over 4 N-subtiles). Block: 4 waves.
// ---------------------------------------------------------------------------
template <int MODE>
__global__ __launch_bounds__(128)
void gemm_bf16(const unsigned short* __restrict__ A,
               const unsigned short* __restrict__ W,
               const float* __restrict__ bias,
               void* __restrict__ Out, int ldo) {
  const int K = 512;
  const int lane = threadIdx.x & 31;
  const int wid  = threadIdx.x >> 5;
  const int hl   = lane >> 4;    // half-wave: 0/1
  const int l16  = lane & 15;

  const int m0 = blockIdx.x * 64 + wid * 16;
  const int n0 = blockIdx.y * 64;

  f32x8 acc[4] = {};
  const unsigned short* Arow = A + (m0 + l16) * K;

  for (int k0 = 0; k0 < K; k0 += 32) {
    ABFrag a;
    a.h[0] = *(const bf16x8*)(Arow + k0 + hl * 8);
    a.h[1] = *(const bf16x8*)(Arow + k0 + 16 + hl * 8);
#pragma unroll
    for (int j = 0; j < 4; ++j) {
      const unsigned short* Wp = W + (n0 + j * 16 + l16) * K + k0 + hl * 8;
      ABFrag b;
      b.h[0] = *(const bf16x8*)(Wp);
      b.h[1] = *(const bf16x8*)(Wp + 16);
      acc[j] = wmma_bf16(a.v, b.v, acc[j]);
    }
  }

#pragma unroll
  for (int j = 0; j < 4; ++j) {
    const int n  = n0 + j * 16 + l16;
    const float bv = bias[n];
#pragma unroll
    for (int i = 0; i < 8; ++i) {
      const int m = m0 + hl * 8 + i;
      const float v = acc[j][i] + bv;
      if (MODE == 0)      ((unsigned short*)Out)[m * ldo + n] = f2bf(v);
      else if (MODE == 1) ((unsigned short*)Out)[n * ldo + m] = f2bf(v);
      else                ((float*)Out)[m * ldo + n] = v;
    }
  }
}

// ---------------------------------------------------------------------------
// Flash attention: per head, O = softmax(Q K^T / 8) V, online softmax.
// One wave owns 16 query rows; KV tiles of 64 keys. Q frags resident.
// P (16x64) staged through per-wave LDS to re-layout C-frag -> A-frag.
// ---------------------------------------------------------------------------
__global__ __launch_bounds__(128)
void flash_attn(const unsigned short* __restrict__ Q,
                const unsigned short* __restrict__ Km,
                const unsigned short* __restrict__ Vt,   // [512][4096]
                unsigned short* __restrict__ Oattn) {
  const int DM = 512, NB = 4096, DK = 64;
  __shared__ __align__(16) unsigned short lp[4][16 * 64];

  const int lane = threadIdx.x & 31;
  const int wid  = threadIdx.x >> 5;
  const int hl   = lane >> 4;
  const int l16  = lane & 15;
  const int h    = blockIdx.y;
  const int m0   = (blockIdx.x * 4 + wid) * 16;
  const float Cs = 0.125f * 1.44269504088896f;   // 1/sqrt(64) * log2(e)

  // Resident Q fragments (d = 0..63 -> 2 A-frags)
  ABFrag qa[2];
  {
    const unsigned short* qp = Q + (m0 + l16) * DM + h * DK;
#pragma unroll
    for (int c = 0; c < 2; ++c) {
      qa[c].h[0] = *(const bf16x8*)(qp + c * 32 + hl * 8);
      qa[c].h[1] = *(const bf16x8*)(qp + c * 32 + 16 + hl * 8);
    }
  }

  f32x8 o[4] = {};
  float rmax[8], rsum[8];
#pragma unroll
  for (int i = 0; i < 8; ++i) { rmax[i] = -3.0e38f; rsum[i] = 0.0f; }

  unsigned short* lpw = lp[wid];

  for (int kv = 0; kv < NB; kv += 64) {
    // ---- S = Q K^T for 64 keys (4 subtiles x 2 K-chunks) ----
    f32x8 s[4] = {};
#pragma unroll
    for (int j = 0; j < 4; ++j) {
      const unsigned short* kp = Km + (kv + j * 16 + l16) * DM + h * DK;
#pragma unroll
      for (int c = 0; c < 2; ++c) {
        ABFrag b;
        b.h[0] = *(const bf16x8*)(kp + c * 32 + hl * 8);
        b.h[1] = *(const bf16x8*)(kp + c * 32 + 16 + hl * 8);
        s[j] = wmma_bf16(qa[c].v, b.v, s[j]);
      }
    }

    if (kv + 64 < NB)  // warm L2/L0 for the next key tile
      __builtin_prefetch(Km + (kv + 64 + lane) * DM + h * DK, 0, 1);

    // ---- online softmax over this 64-key tile ----
#pragma unroll
    for (int i = 0; i < 8; ++i) {
      float tm = fmaxf(fmaxf(s[0][i], s[1][i]), fmaxf(s[2][i], s[3][i]));
      tm = fmaxf(tm, __shfl_xor(tm, 1));
      tm = fmaxf(tm, __shfl_xor(tm, 2));
      tm = fmaxf(tm, __shfl_xor(tm, 4));
      tm = fmaxf(tm, __shfl_xor(tm, 8));
      const float nm    = fmaxf(rmax[i], tm);
      const float alpha = __builtin_exp2f((rmax[i] - nm) * Cs);
      const float nmC   = nm * Cs;
      float ps = 0.0f;
#pragma unroll
      for (int j = 0; j < 4; ++j) {
        const float p = __builtin_exp2f(s[j][i] * Cs - nmC);
        s[j][i] = p;
        ps += p;
      }
      ps += __shfl_xor(ps, 1);
      ps += __shfl_xor(ps, 2);
      ps += __shfl_xor(ps, 4);
      ps += __shfl_xor(ps, 8);
      rsum[i] = rsum[i] * alpha + ps;
      rmax[i] = nm;
#pragma unroll
      for (int j = 0; j < 4; ++j) o[j][i] *= alpha;
    }

    // ---- re-layout P: C-frag -> LDS -> A-frag (bf16) ----
#pragma unroll
    for (int j = 0; j < 4; ++j)
#pragma unroll
      for (int i = 0; i < 8; ++i)
        lpw[(hl * 8 + i) * 64 + j * 16 + l16] = f2bf(s[j][i]);
    asm volatile("s_wait_dscnt 0" ::: "memory");

    ABFrag pa[2];
#pragma unroll
    for (int c = 0; c < 2; ++c) {
      pa[c].h[0] = *(const bf16x8*)(lpw + l16 * 64 + c * 32 + hl * 8);
      pa[c].h[1] = *(const bf16x8*)(lpw + l16 * 64 + c * 32 + 16 + hl * 8);
    }

    // ---- O += P V  (V^T rows are d, contiguous along keys) ----
#pragma unroll
    for (int j = 0; j < 4; ++j) {
      const unsigned short* vp = Vt + (h * DK + j * 16 + l16) * NB + kv;
#pragma unroll
      for (int c = 0; c < 2; ++c) {
        ABFrag b;
        b.h[0] = *(const bf16x8*)(vp + c * 32 + hl * 8);
        b.h[1] = *(const bf16x8*)(vp + c * 32 + 16 + hl * 8);
        o[j] = wmma_bf16(pa[c].v, b.v, o[j]);
      }
    }
  }

  // ---- normalize and store attn (bf16, [4096 x 512]) ----
#pragma unroll
  for (int i = 0; i < 8; ++i) {
    const float inv = 1.0f / rsum[i];
    const int m = m0 + hl * 8 + i;
#pragma unroll
    for (int j = 0; j < 4; ++j)
      Oattn[m * DM + h * DK + j * 16 + l16] = f2bf(o[j][i] * inv);
  }
}

// ---------------------------------------------------------------------------
// Host-side orchestration
// ---------------------------------------------------------------------------
extern "C" void kernel_launch(void* const* d_in, const int* in_sizes, int n_in,
                              void* d_out, int out_size, void* d_ws, size_t ws_size,
                              hipStream_t stream) {
  const int B = 4096, DM = 512;
  const float* x  = (const float*)d_in[0];
  const float* Wq = (const float*)d_in[1];
  const float* bq = (const float*)d_in[2];
  const float* Wk = (const float*)d_in[3];
  const float* bk = (const float*)d_in[4];
  const float* Wv = (const float*)d_in[5];
  const float* bv = (const float*)d_in[6];
  const float* Wo = (const float*)d_in[7];
  const float* bo = (const float*)d_in[8];

  const size_t SX = (size_t)B * DM;     // 2M elems
  const size_t SW = (size_t)DM * DM;    // 256K elems
  unsigned short* ws  = (unsigned short*)d_ws;
  unsigned short* xb  = ws;
  unsigned short* wqb = xb + SX;
  unsigned short* wkb = wqb + SW;
  unsigned short* wvb = wkb + SW;
  unsigned short* wob = wvb + SW;
  unsigned short* qb  = wob + SW;
  unsigned short* kb  = qb + SX;
  unsigned short* vtb = kb + SX;       // [512][4096] = V^T
  unsigned short* ab  = vtb + SX;      // attn result bf16

  auto cvt = [&](const float* in, unsigned short* out, size_t n) {
    int blocks = (int)((n / 4 + 255) / 256);
    cvt_bf16<<<blocks, 256, 0, stream>>>(in, out, (int)n);
  };
  cvt(x,  xb,  SX);
  cvt(Wq, wqb, SW);
  cvt(Wk, wkb, SW);
  cvt(Wv, wvb, SW);
  cvt(Wo, wob, SW);

  dim3 gg(B / 64, DM / 64), bb(128);
  gemm_bf16<0><<<gg, bb, 0, stream>>>(xb, wqb, bq, qb,  DM);   // Q
  gemm_bf16<0><<<gg, bb, 0, stream>>>(xb, wkb, bk, kb,  DM);   // K
  gemm_bf16<1><<<gg, bb, 0, stream>>>(xb, wvb, bv, vtb, B);    // V^T

  flash_attn<<<dim3(B / 64, 8), 128, 0, stream>>>(qb, kb, vtb, ab);

  gemm_bf16<2><<<gg, bb, 0, stream>>>(ab, wob, bo, d_out, DM); // out proj
}
